// AttentionBlock_12567074308944
// MI455X (gfx1250) — matrix operand order repaired and checked by reading, not compiled, not run on hardware
//
#include <hip/hip_runtime.h>

// ---------------------------------------------------------------------------
// AttentionBlock for MI455X (gfx1250): GN -> qkv GEMM -> QK^T -> softmax ->
// PV -> proj GEMM + residual.  All GEMMs via v_wmma_f32_16x16x32_bf16
// (bf16 inputs, f32 accumulate).  B=16, C=512, N=H*W=1024.
// Block tile 128x256 (8 waves, 64x64 per wave, 4x4 WMMA accum tiles).
// ---------------------------------------------------------------------------

typedef __attribute__((ext_vector_type(16))) __bf16 v16bf;
typedef __attribute__((ext_vector_type(8)))  __bf16 v8bf;
typedef __attribute__((ext_vector_type(8)))  float  v8f;

#define BM 128
#define BN 256
#define BK 32
#define SA 40    // padded LDS row stride (bf16 elems) for A tile (32+8)
#define SB 264   // padded LDS row stride (bf16 elems) for B tile (256+8)

__device__ __forceinline__ unsigned short f2bf(float f) {
  unsigned int u = __float_as_uint(f);
  u += 0x7FFFu + ((u >> 16) & 1u);         // round-to-nearest-even
  return (unsigned short)(u >> 16);
}

__device__ __forceinline__ v16bf frag16(const unsigned short* p0,
                                        const unsigned short* p1) {
  v8bf lo = *(const v8bf*)p0;
  v8bf hi = *(const v8bf*)p1;
  return __builtin_shufflevector(lo, hi, 0,1,2,3,4,5,6,7,8,9,10,11,12,13,14,15);
}

// A is [M][K] row-major bf16 (lda elems), B is [K][N] row-major bf16 (ldb).
// 8 waves arranged 2(M) x 4(N); each wave computes a 64x64 tile = 4x4 WMMAs.
__device__ __forceinline__ void gemm_core(
    const unsigned short* __restrict__ A, int lda,
    const unsigned short* __restrict__ B, int ldb,
    int K, int mBlock, int nBlock, v8f acc[4][4])
{
  __shared__ unsigned short As[BM * SA];
  __shared__ unsigned short Bs[BK * SB];
  const int t    = threadIdx.x;
  const int lane = t & 31;
  const int wid  = t >> 5;
  const int half = lane >> 4;       // 0: lanes 0-15, 1: lanes 16-31
  const int ln   = lane & 15;
  const int wm   = (wid >> 2) * 64; // wave M offset in block tile
  const int wn   = (wid & 3) * 64;  // wave N offset in block tile
  const int kf   = half * 8;        // A-frag K sub-offset per ISA layout
  const int ar = t >> 2, ac = (t & 3) << 3;  // A copy: 2x(64 rows x 32 cols)
  const int br = t >> 3, bc = (t & 7) << 5;  // B copy: 32 rows x 256 cols

  #pragma unroll
  for (int i = 0; i < 4; ++i)
    #pragma unroll
    for (int j = 0; j < 4; ++j)
      #pragma unroll
      for (int r = 0; r < 8; ++r) acc[i][j][r] = 0.0f;

  for (int k0 = 0; k0 < K; k0 += BK) {
    // coalesced global -> LDS (A: 2 uint4/thread, B: 4 uint4/thread)
    *(uint4*)&As[ar * SA + ac] =
        *(const uint4*)(A + (size_t)(mBlock + ar) * lda + (k0 + ac));
    *(uint4*)&As[(ar + 64) * SA + ac] =
        *(const uint4*)(A + (size_t)(mBlock + ar + 64) * lda + (k0 + ac));
    const uint4* bsrc =
        (const uint4*)(B + (size_t)(k0 + br) * ldb + (nBlock + bc));
    *(uint4*)&Bs[br * SB + bc]      = bsrc[0];
    *(uint4*)&Bs[br * SB + bc + 8]  = bsrc[1];
    *(uint4*)&Bs[br * SB + bc + 16] = bsrc[2];
    *(uint4*)&Bs[br * SB + bc + 24] = bsrc[3];
    __syncthreads();

    v16bf af[4], bfr[4];
    #pragma unroll
    for (int i = 0; i < 4; ++i) {
      const int row = (wm + i * 16 + ln) * SA;
      af[i] = frag16(&As[row + kf], &As[row + 16 + kf]);
    }
    #pragma unroll
    for (int j = 0; j < 4; ++j) {
      const int col = wn + j * 16 + ln;
      #pragma unroll
      for (int s = 0; s < 16; ++s) {
        const int kk = (s < 8) ? (kf + s) : (16 + kf + (s - 8));
        bfr[j][s] = __builtin_bit_cast(__bf16, Bs[kk * SB + col]);
      }
    }
    #pragma unroll
    for (int i = 0; i < 4; ++i)
      #pragma unroll
      for (int j = 0; j < 4; ++j)
        acc[i][j] = __builtin_amdgcn_wmma_f32_16x16x32_bf16(
            false, af[i], false, bfr[j], (short)0, acc[i][j], false, false);
    __syncthreads();
  }
}

// -------------------------- weight conversion ------------------------------
__global__ void convert_weights(const float* __restrict__ qkv_w,
                                const float* __restrict__ proj_w,
                                unsigned short* __restrict__ wq,
                                unsigned short* __restrict__ wp) {
  int i = blockIdx.x * 256 + threadIdx.x;
  if (i < 1536 * 512) wq[i] = f2bf(qkv_w[i]);
  if (i < 512 * 512)  wp[i] = f2bf(proj_w[i]);
}

// ------------------------------ group norm ---------------------------------
// one block per (batch, group): 16 channels x 1024 pixels contiguous
__global__ void gn_kernel(const float* __restrict__ x,
                          const float* __restrict__ w,
                          const float* __restrict__ b,
                          unsigned short* __restrict__ xn) {
  const int bb = blockIdx.x >> 5;
  const int g  = blockIdx.x & 31;
  const int t  = threadIdx.x;
  const float* xp = x + ((size_t)bb * 512 + g * 16) * 1024;
  float s = 0.f, s2 = 0.f;
  for (int i = t; i < 16384; i += 256) { float v = xp[i]; s += v; s2 += v * v; }
  __shared__ float r1[256], r2[256];
  r1[t] = s; r2[t] = s2; __syncthreads();
  for (int st = 128; st > 0; st >>= 1) {
    if (t < st) { r1[t] += r1[t + st]; r2[t] += r2[t + st]; }
    __syncthreads();
  }
  const float mean = r1[0] * (1.f / 16384.f);
  const float var  = r2[0] * (1.f / 16384.f) - mean * mean;
  const float rstd = rsqrtf(var + 1e-5f);
  unsigned short* op = xn + ((size_t)bb * 512 + g * 16) * 1024;
  for (int i = t; i < 16384; i += 256) {
    int c = g * 16 + (i >> 10);
    op[i] = f2bf((xp[i] - mean) * rstd * w[c] + b[c]);
  }
}

// ------------------------------ qkv GEMM -----------------------------------
__global__ void qkv_gemm(const unsigned short* __restrict__ wq,
                         const unsigned short* __restrict__ xn,
                         const float* __restrict__ qkv_b,
                         unsigned short* __restrict__ qT,
                         unsigned short* __restrict__ kk,
                         unsigned short* __restrict__ vT) {
  const int bb = blockIdx.z;
  const int mBlock = blockIdx.y * BM, nBlock = blockIdx.x * BN;
  v8f acc[4][4];
  gemm_core(wq, 512, xn + (size_t)bb * 512 * 1024, 1024, 512, mBlock, nBlock, acc);
  const int lane = threadIdx.x & 31, wid = threadIdx.x >> 5;
  const int half = lane >> 4, ln = lane & 15;
  const int wm = (wid >> 2) * 64, wn = (wid & 3) * 64;
  #pragma unroll
  for (int i = 0; i < 4; ++i)
    #pragma unroll
    for (int j = 0; j < 4; ++j)
      #pragma unroll
      for (int r = 0; r < 8; ++r) {
        int m = mBlock + wm + i * 16 + half * 8 + r;
        int n = nBlock + wn + j * 16 + ln;
        float v = acc[i][j][r] + qkv_b[m];
        if (m < 512)            // q, scaled, stored transposed [n][c]
          qT[((size_t)bb * 1024 + n) * 512 + m] = f2bf(v * 0.125f);
        else if (m < 1024)      // k stored [c][m]
          kk[((size_t)bb * 512 + (m - 512)) * 1024 + n] = f2bf(v);
        else                    // v stored transposed [m][c]
          vT[((size_t)bb * 1024 + n) * 512 + (m - 1024)] = f2bf(v);
      }
}

// --------------------------- S = (q*scale)^T k -----------------------------
__global__ void attn_gemm(const unsigned short* __restrict__ qT,
                          const unsigned short* __restrict__ kk,
                          float* __restrict__ S, int b0) {
  const int slot = blockIdx.z;
  const int bb = b0 + slot;
  const int mBlock = blockIdx.y * BM, nBlock = blockIdx.x * BN;
  v8f acc[4][4];
  gemm_core(qT + (size_t)bb * 1024 * 512, 512,
            kk + (size_t)bb * 512 * 1024, 1024, 512, mBlock, nBlock, acc);
  float* Sp = S + (size_t)slot * 1024 * 1024;
  const int lane = threadIdx.x & 31, wid = threadIdx.x >> 5;
  const int half = lane >> 4, ln = lane & 15;
  const int wm = (wid >> 2) * 64, wn = (wid & 3) * 64;
  #pragma unroll
  for (int i = 0; i < 4; ++i)
    #pragma unroll
    for (int j = 0; j < 4; ++j)
      #pragma unroll
      for (int r = 0; r < 8; ++r) {
        int m = mBlock + wm + i * 16 + half * 8 + r;
        int n = nBlock + wn + j * 16 + ln;
        Sp[(size_t)m * 1024 + n] = acc[i][j][r];
      }
}

// ------------------------------ softmax ------------------------------------
// one block per row; writes bf16 P in-place over the first half of its own
// f32 row (all reads complete before the first barrier, so this is safe).
__global__ void softmax_rows(float* __restrict__ S) {
  const int slot = blockIdx.x >> 10;
  const int row  = blockIdx.x & 1023;
  const int t = threadIdx.x;
  float* Sp = S + (size_t)slot * 1024 * 1024;
  const float* rp = Sp + (size_t)row * 1024;
  float v[4];
  float mx = -__builtin_inff();
  #pragma unroll
  for (int i = 0; i < 4; ++i) { v[i] = rp[t + 256 * i]; mx = fmaxf(mx, v[i]); }
  __shared__ float red[256];
  red[t] = mx; __syncthreads();
  for (int s = 128; s > 0; s >>= 1) {
    if (t < s) red[t] = fmaxf(red[t], red[t + s]);
    __syncthreads();
  }
  mx = red[0]; __syncthreads();
  float sum = 0.f;
  #pragma unroll
  for (int i = 0; i < 4; ++i) { v[i] = __expf(v[i] - mx); sum += v[i]; }
  red[t] = sum; __syncthreads();
  for (int s = 128; s > 0; s >>= 1) {
    if (t < s) red[t] += red[t + s];
    __syncthreads();
  }
  const float inv = 1.0f / red[0];
  unsigned short* P = (unsigned short*)Sp + (size_t)row * 2048;  // lda=2048 u16
  #pragma unroll
  for (int i = 0; i < 4; ++i) P[t + 256 * i] = f2bf(v[i] * inv);
}

// ------------------------------ O = P V^T ----------------------------------
// O2[n_query][c] = sum_m P[n][m] vT[m][c]; epilogue re-transposes to [c][n]
__global__ void out_gemm(const unsigned short* __restrict__ P,   // base of S
                         const unsigned short* __restrict__ vT,
                         unsigned short* __restrict__ Ob, int b0) {
  const int slot = blockIdx.z;
  const int bb = b0 + slot;
  const unsigned short* Pp = P + (size_t)slot * 1024 * 2048;     // lda=2048
  const int mBlock = blockIdx.y * BM, nBlock = blockIdx.x * BN;
  v8f acc[4][4];
  gemm_core(Pp, 2048, vT + (size_t)bb * 1024 * 512, 512, 1024, mBlock, nBlock, acc);
  const int lane = threadIdx.x & 31, wid = threadIdx.x >> 5;
  const int half = lane >> 4, ln = lane & 15;
  const int wm = (wid >> 2) * 64, wn = (wid & 3) * 64;
  #pragma unroll
  for (int i = 0; i < 4; ++i)
    #pragma unroll
    for (int j = 0; j < 4; ++j)
      #pragma unroll
      for (int r = 0; r < 8; ++r) {
        int m = mBlock + wm + i * 16 + half * 8 + r;   // query position
        int n = nBlock + wn + j * 16 + ln;             // channel
        Ob[((size_t)bb * 512 + n) * 1024 + m] = f2bf(acc[i][j][r]);
      }
}

// ----------------------- proj GEMM + bias + residual -----------------------
__global__ void proj_gemm(const unsigned short* __restrict__ wp,
                          const unsigned short* __restrict__ Ob,
                          const float* __restrict__ proj_b,
                          const float* __restrict__ x,
                          float* __restrict__ out) {
  const int bb = blockIdx.z;
  const int mBlock = blockIdx.y * BM, nBlock = blockIdx.x * BN;
  v8f acc[4][4];
  gemm_core(wp, 512, Ob + (size_t)bb * 512 * 1024, 1024, 512, mBlock, nBlock, acc);
  const int lane = threadIdx.x & 31, wid = threadIdx.x >> 5;
  const int half = lane >> 4, ln = lane & 15;
  const int wm = (wid >> 2) * 64, wn = (wid & 3) * 64;
  #pragma unroll
  for (int i = 0; i < 4; ++i)
    #pragma unroll
    for (int j = 0; j < 4; ++j)
      #pragma unroll
      for (int r = 0; r < 8; ++r) {
        int m = mBlock + wm + i * 16 + half * 8 + r;   // out channel
        int n = nBlock + wn + j * 16 + ln;             // spatial
        size_t idx = ((size_t)bb * 512 + m) * 1024 + n;
        out[idx] = x[idx] + proj_b[m] + acc[i][j][r];
      }
}

// ---------------------------------------------------------------------------
extern "C" void kernel_launch(void* const* d_in, const int* in_sizes, int n_in,
                              void* d_out, int out_size, void* d_ws, size_t ws_size,
                              hipStream_t stream) {
  (void)in_sizes; (void)n_in; (void)out_size;
  const float* x      = (const float*)d_in[0];
  const float* gn_w   = (const float*)d_in[1];
  const float* gn_b   = (const float*)d_in[2];
  const float* qkv_w  = (const float*)d_in[3];
  const float* qkv_b  = (const float*)d_in[4];
  const float* proj_w = (const float*)d_in[5];
  const float* proj_b = (const float*)d_in[6];
  float* out = (float*)d_out;
  char*  ws  = (char*)d_ws;

  size_t off = 0;
  auto take = [&](size_t bytes) {
    size_t r = off; off += (bytes + 255) & ~(size_t)255; return r;
  };
  unsigned short* wq = (unsigned short*)(ws + take((size_t)1536 * 512 * 2));
  unsigned short* wp = (unsigned short*)(ws + take((size_t)512 * 512 * 2));
  unsigned short* xn = (unsigned short*)(ws + take((size_t)16 * 512 * 1024 * 2));
  unsigned short* qT = (unsigned short*)(ws + take((size_t)16 * 1024 * 512 * 2));
  unsigned short* kk = (unsigned short*)(ws + take((size_t)16 * 512 * 1024 * 2));
  unsigned short* vT = (unsigned short*)(ws + take((size_t)16 * 1024 * 512 * 2));
  unsigned short* Ob = (unsigned short*)(ws + take((size_t)16 * 512 * 1024 * 2));
  // S gets whatever is left: 4MB per in-flight batch (1..16), deterministic
  // since ws_size is fixed for the session.
  const size_t sbytes = (size_t)1024 * 1024 * 4;
  float* S = (float*)(ws + off);
  int nb = (ws_size > off) ? (int)((ws_size - off) / sbytes) : 1;
  if (nb < 1) nb = 1;
  if (nb > 16) nb = 16;
  while (16 % nb) --nb;   // chunk must divide 16: 16,8,4,2,1

  convert_weights<<<(1536 * 512 + 255) / 256, 256, 0, stream>>>(qkv_w, proj_w, wq, wp);
  gn_kernel<<<512, 256, 0, stream>>>(x, gn_w, gn_b, xn);
  qkv_gemm<<<dim3(1024 / BN, 1536 / BM, 16), 256, 0, stream>>>(wq, xn, qkv_b, qT, kk, vT);

  // attention in batch chunks; score matrices stay L2-resident
  for (int b0 = 0; b0 < 16; b0 += nb) {
    attn_gemm<<<dim3(1024 / BN, 1024 / BM, nb), 256, 0, stream>>>(qT, kk, S, b0);
    softmax_rows<<<1024 * nb, 256, 0, stream>>>(S);
    out_gemm<<<dim3(512 / BN, 1024 / BM, nb), 256, 0, stream>>>(
        (const unsigned short*)S, vT, Ob, b0);
  }

  proj_gemm<<<dim3(1024 / BN, 512 / BM, 16), 256, 0, stream>>>(wp, Ob, proj_b, x, out);
}
